// AdditiveAlignment_22101901705580
// MI455X (gfx1250) — compile-verified, matrix operand order checked by reading, not run on hardware
//
#include <hip/hip_runtime.h>
#include <hip/hip_bf16.h>
#include <math.h>

#define B_ 64
#define S_ 2048
#define H_ 512
#define A_ 512

typedef __attribute__((ext_vector_type(16))) __bf16 v16bf;
typedef __attribute__((ext_vector_type(8)))  float  v8f;
typedef __attribute__((ext_vector_type(4)))  int    v4i;

#define GLOBAL_AS __attribute__((address_space(1)))
#define LDS_AS    __attribute__((address_space(3)))

#if defined(__has_builtin)
#  if __has_builtin(__builtin_amdgcn_global_load_async_to_lds_b128)
#    define HAVE_ASYNC_LDS 1
#  endif
#endif
#ifndef HAVE_ASYNC_LDS
#  define HAVE_ASYNC_LDS 0
#endif

union BF16x16 {
    v16bf v;
    unsigned short u[16];
};

// round-to-nearest-even fp32 -> bf16 (bit pattern)
__device__ __forceinline__ unsigned short f2bf(float f) {
    unsigned int x = __float_as_uint(f);
    x += 0x7FFFu + ((x >> 16) & 1u);
    return (unsigned short)(x >> 16);
}

__device__ __forceinline__ void wait_async0() {
#if HAVE_ASYNC_LDS
#  if __has_builtin(__builtin_amdgcn_s_wait_asynccnt)
    __builtin_amdgcn_s_wait_asynccnt(0);
#  else
    asm volatile("s_wait_asynccnt 0x0" ::: "memory");
#  endif
#endif
}

// Stage one 16-column W_enc strip (16 x 512 bf16 = 16 KB) into an LDS buffer,
// layout [n][k].  Async (fire-and-forget, ASYNCcnt) when available.
__device__ __forceinline__ void prefetch_tile(
        const unsigned short* __restrict__ Wb, int t,
        unsigned short* dstBase, int tid) {
#if HAVE_ASYNC_LDS
#pragma unroll
    for (int r = 0; r < 4; ++r) {
        const int e = (r * 256 + tid) * 8;        // ushort index, 16B chunks
        const int n = e >> 9;
        const int k = e & (H_ - 1);
        const unsigned short* g = Wb + (t * 16 + n) * H_ + k;
        __builtin_amdgcn_global_load_async_to_lds_b128(
            (GLOBAL_AS v4i*)(v4i*)g,
            (LDS_AS v4i*)(v4i*)(dstBase + e),
            0, 0);
    }
#else
    for (int idx = tid; idx < 16 * H_; idx += 256) {
        int n = idx >> 9;
        int k = idx & (H_ - 1);
        dstBase[idx] = Wb[(t * 16 + n) * H_ + k];
    }
#endif
}

// ---------------- kernel 0: W_enc fp32 -> bf16 ----------------
__global__ __launch_bounds__(256) void convert_w_kernel(
        const float* __restrict__ W, unsigned short* __restrict__ Wb) {
    int idx = blockIdx.x * 256 + threadIdx.x;   // < A_*H_
    Wb[idx] = f2bf(W[idx]);
}

// ---------------- kernel 1: decbias[b,a] = dec.Wdec[a] + b_dec[a] + b_enc[a] ----
__global__ __launch_bounds__(256) void decbias_kernel(
        const float* __restrict__ dec, const float* __restrict__ Wd,
        const float* __restrict__ bd, const float* __restrict__ be,
        float* __restrict__ out) {
    int idx = blockIdx.x * 256 + threadIdx.x;   // < B_*A_
    int b = idx >> 9;
    int a = idx & (A_ - 1);
    const float* d = dec + b * H_;
    const float* w = Wd + a * H_;
    float acc = 0.f;
#pragma unroll 8
    for (int k = 0; k < H_; ++k) acc += d[k] * w[k];
    out[idx] = acc + bd[a] + be[a];
}

// ---------------- kernel 2: fused enc-proj GEMM + tanh + V-dot -> scores -------
// grid: (B_*S_)/128 blocks of 256 threads (8 waves). Wave w handles 16 rows.
__global__ __launch_bounds__(256, 1) void enc_score_kernel(
        const float* __restrict__ enc,            // (B,S,H) fp32
        const unsigned short* __restrict__ Wb,    // (A,H) bf16 bits
        const float* __restrict__ decbias,        // (B,A)
        const float* __restrict__ Vvec,           // (A)
        float* __restrict__ scores) {             // (B,S)
    __shared__ __align__(16) unsigned short ldsW[2][16 * H_];  // double buffer, 32 KB

    const int tid  = threadIdx.x;
    const int wave = tid >> 5;
    const int lane = tid & 31;
    const int half = lane >> 4;      // 0: C rows 0-7 / K low sub-block; 1: rows 8-15
    const int l15  = lane & 15;
    const int khoff = half ? 8 : 0;

    const long rowBase = (long)blockIdx.x * 128 + wave * 16;   // flat (b*S + s)
    const int  b = (int)(rowBase / S_);

    // kick off async prefetch of the first W strip
    prefetch_tile(Wb, 0, &ldsW[0][0], tid);

    // ---- load A fragments once (encoder rows, fp32 -> bf16), 16 K-steps ----
    BF16x16 afrag[16];
    {
        const float* src = enc + (rowBase + l15) * (long)H_;
#pragma unroll
        for (int ks = 0; ks < 16; ++ks) {
            const int base1 = ks * 32 + khoff;            // K: base1..base1+7
            const int base2 = base1 + 16;                 // K: base2..base2+7
            float4 p0 = *(const float4*)(src + base1);
            float4 p1 = *(const float4*)(src + base1 + 4);
            float4 p2 = *(const float4*)(src + base2);
            float4 p3 = *(const float4*)(src + base2 + 4);
            afrag[ks].u[0]  = f2bf(p0.x); afrag[ks].u[1]  = f2bf(p0.y);
            afrag[ks].u[2]  = f2bf(p0.z); afrag[ks].u[3]  = f2bf(p0.w);
            afrag[ks].u[4]  = f2bf(p1.x); afrag[ks].u[5]  = f2bf(p1.y);
            afrag[ks].u[6]  = f2bf(p1.z); afrag[ks].u[7]  = f2bf(p1.w);
            afrag[ks].u[8]  = f2bf(p2.x); afrag[ks].u[9]  = f2bf(p2.y);
            afrag[ks].u[10] = f2bf(p2.z); afrag[ks].u[11] = f2bf(p2.w);
            afrag[ks].u[12] = f2bf(p3.x); afrag[ks].u[13] = f2bf(p3.y);
            afrag[ks].u[14] = f2bf(p3.z); afrag[ks].u[15] = f2bf(p3.w);
        }
    }

    float partial[8];
#pragma unroll
    for (int j = 0; j < 8; ++j) partial[j] = 0.f;

#define LOADB(dst, ksv)                                                        \
    {                                                                          \
        const int bb = (ksv) * 32 + khoff;                                     \
        *(uint2*)&(dst).u[0]  = *(const uint2*)(wp + bb);                      \
        *(uint2*)&(dst).u[4]  = *(const uint2*)(wp + bb + 4);                  \
        *(uint2*)&(dst).u[8]  = *(const uint2*)(wp + bb + 16);                 \
        *(uint2*)&(dst).u[12] = *(const uint2*)(wp + bb + 20);                 \
    }

    // ---- loop over 32 N-tiles of 16 columns of A_ ----
    for (int t = 0; t < 32; ++t) {
        const int cur = t & 1;
        wait_async0();        // this wave's async loads into ldsW[cur] done
        __syncthreads();      // everyone's loads visible; buf cur^1 free
        if (t + 1 < 32)
            prefetch_tile(Wb, t + 1, &ldsW[cur ^ 1][0], tid);

        const unsigned short* wp = &ldsW[cur][0] + l15 * H_;  // column n = l15

        v8f c = {};
        BF16x16 bf0, bf1;
        LOADB(bf0, 0);
#pragma unroll
        for (int ks = 0; ks < 16; ++ks) {
            if (ks + 1 < 16) {                 // prefetch next B frag (1 deep)
                if (ks & 1) LOADB(bf0, ks + 1) else LOADB(bf1, ks + 1)
            }
            c = __builtin_amdgcn_wmma_f32_16x16x32_bf16(
                    false, afrag[ks].v, false, (ks & 1) ? bf1.v : bf0.v,
                    (short)0, c, false, false);
        }

        // fused epilogue: h = tanh(c + decbias[b,col]); partial += h * V[col]
        const int col = t * 16 + l15;
        const float db = decbias[b * A_ + col];
        const float vv = Vvec[col];
#pragma unroll
        for (int j = 0; j < 8; ++j)
            partial[j] += tanhf(c[j] + db) * vv;
    }
#undef LOADB

    // reduce across the 16 lanes of each half (xor masks stay within half)
#pragma unroll
    for (int m = 1; m < 16; m <<= 1) {
#pragma unroll
        for (int j = 0; j < 8; ++j)
            partial[j] += __shfl_xor(partial[j], m, 32);
    }
    if (l15 == 0) {
        long s0 = rowBase + half * 8;    // lanes<16 -> rows 0-7, lanes>=16 -> rows 8-15
#pragma unroll
        for (int j = 0; j < 8; ++j) scores[s0 + j] = partial[j];
    }
}

// ---------------- kernel 3: softmax over S per batch row (in place) -----------
__global__ __launch_bounds__(256) void softmax_kernel(float* __restrict__ scores) {
    const int b = blockIdx.x;
    const int tid = threadIdx.x;
    float* row = scores + (long)b * S_;
    __shared__ float red[256];

    float m = -INFINITY;
    for (int i = tid; i < S_; i += 256) m = fmaxf(m, row[i]);
    red[tid] = m;
    __syncthreads();
    for (int s = 128; s > 0; s >>= 1) {
        if (tid < s) red[tid] = fmaxf(red[tid], red[tid + s]);
        __syncthreads();
    }
    m = red[0];
    __syncthreads();

    float sum = 0.f;
    for (int i = tid; i < S_; i += 256) {
        float e = __expf(row[i] - m);
        row[i] = e;
        sum += e;
    }
    red[tid] = sum;
    __syncthreads();
    for (int s = 128; s > 0; s >>= 1) {
        if (tid < s) red[tid] += red[tid + s];
        __syncthreads();
    }
    const float inv = 1.0f / red[0];
    for (int i = tid; i < S_; i += 256) row[i] *= inv;
}

// ---------------- kernel 4: ctx = attn @ enc ; out = concat(ctx, dec) ---------
__global__ __launch_bounds__(256) void ctx_kernel(
        const float* __restrict__ enc, const float* __restrict__ attn,
        const float* __restrict__ dec, float* __restrict__ out) {
    const int b = blockIdx.x;
    const int tid = threadIdx.x;
    __shared__ float at[S_];   // 8 KB
    for (int i = tid; i < S_; i += 256) at[i] = attn[(long)b * S_ + i];
    __syncthreads();

    const float* eb = enc + (long)b * S_ * H_;
    for (int h = tid; h < H_; h += 256) {
        float acc = 0.f;
#pragma unroll 4
        for (int s = 0; s < S_; ++s) acc += at[s] * eb[(long)s * H_ + h];
        out[(long)b * (2 * H_) + h] = acc;
    }
    for (int h = tid; h < H_; h += 256)
        out[(long)b * (2 * H_) + H_ + h] = dec[b * H_ + h];
}

extern "C" void kernel_launch(void* const* d_in, const int* in_sizes, int n_in,
                              void* d_out, int out_size, void* d_ws, size_t ws_size,
                              hipStream_t stream) {
    (void)in_sizes; (void)n_in; (void)out_size; (void)ws_size;
    const float* dec = (const float*)d_in[0];   // (B,H)
    const float* enc = (const float*)d_in[1];   // (B,S,H)
    const float* Wd  = (const float*)d_in[2];   // (A,H)
    const float* bd  = (const float*)d_in[3];   // (A)
    const float* We  = (const float*)d_in[4];   // (A,H)
    const float* be  = (const float*)d_in[5];   // (A)
    const float* V   = (const float*)d_in[6];   // (1,A)
    // d_in[7] = b_v: constant shift before softmax -> softmax-invariant, unused.
    float* out = (float*)d_out;                 // (B, 2H)

    char* ws = (char*)d_ws;
    unsigned short* Wb = (unsigned short*)ws;                       // 512 KB
    float* decbias = (float*)(ws + (size_t)A_ * H_ * 2);            // 128 KB
    float* scores  = (float*)(ws + (size_t)A_ * H_ * 2 + (size_t)B_ * A_ * 4); // 512 KB

    convert_w_kernel<<<(A_ * H_) / 256, 256, 0, stream>>>(We, Wb);
    decbias_kernel<<<(B_ * A_) / 256, 256, 0, stream>>>(dec, Wd, bd, be, decbias);
    enc_score_kernel<<<(B_ * S_) / 128, 256, 0, stream>>>(enc, Wb, decbias, V, scores);
    softmax_kernel<<<B_, 256, 0, stream>>>(scores);
    ctx_kernel<<<B_, 256, 0, stream>>>(enc, scores, dec, out);
}
